// _PipelineForProfiling_3307124818734
// MI455X (gfx1250) — matrix operand order repaired and checked by reading, not compiled, not run on hardware
//
#include <hip/hip_runtime.h>
#include <hip/hip_bf16.h>

typedef __attribute__((ext_vector_type(16))) _Float16 v16h;
typedef __attribute__((ext_vector_type(8)))  _Float16 v8h;
typedef __attribute__((ext_vector_type(8)))  float    v8f;
typedef unsigned int u32x4 __attribute__((ext_vector_type(4)));
typedef int          i32x8 __attribute__((ext_vector_type(8)));
typedef int          i32x4 __attribute__((ext_vector_type(4)));

// Problem constants
#define B_    2
#define CI_   3
#define CF_   384
#define CQK_  128
#define T_    4
#define H_    128
#define W_    128
#define HL_   32
#define WL_   32
#define NH_   4
#define WT_   2
#define WLO_  8          // lo-res spatial window
#define WSH_  32         // hi-res spatial window
#define DQK_  32         // Cqk / nh
#define DV_   96         // Cf / nh
#define NWIN_ 32         // (T/wt)*(H/ws)*(W/ws) = 2*4*4
#define QROWS_ 2048      // wt*ws*ws
#define KKEYS_ 128       // wt*wl*wl
#define NPIX_  131072    // B*T*H*W
#define THW_   65536     // T*H*W

// Workspace layout (bytes)
#define OFF_WOH  0u
#define OFF_KW   294912u
#define OFF_VWT  2392064u
#define OFF_X    8683520u

static __device__ __forceinline__ v8f v8f_zero() {
  v8f z = {0.f,0.f,0.f,0.f,0.f,0.f,0.f,0.f};
  return z;
}

// ---------------------------------------------------------------------------
// TDM: issue a 2-D f16 tile DMA (tile_d0 x tile_d1 elements) from a row-major
// tensor (row length tensor_d0, row stride stride0, tensor_d1 rows) into LDS.
// D# layout per CDNA5 ISA 8.3/8.4. Wave-level op, TENSORcnt tracked.
// ---------------------------------------------------------------------------
static __device__ __forceinline__ void tdm_load_tile_f16(
    const _Float16* gsrc, unsigned lds_off,
    unsigned tile_d0, unsigned tile_d1,
    unsigned tensor_d0, unsigned tensor_d1, unsigned stride0)
{
  unsigned long long ga = (unsigned long long)(uintptr_t)gsrc;
  u32x4 g0;
  g0[0] = 1u;                                         // count=1 (valid), user mode
  g0[1] = lds_off;                                    // lds_addr (bytes)
  g0[2] = (unsigned)(ga & 0xFFFFFFFFull);             // global_addr[31:0]
  g0[3] = (unsigned)((ga >> 32) & 0x01FFFFFFull)      // global_addr[56:32]
        | (2u << 30);                                 // type = 2 ("image")
  i32x8 g1;
  g1[0] = (int)(1u << 16);                            // wg_mask=0, data_size=1 (2B)
  g1[1] = (int)((tensor_d0 & 0xFFFFu) << 16);         // tensor_dim0[15:0]
  g1[2] = (int)(((tensor_d0 >> 16) & 0xFFFFu)
        | ((tensor_d1 & 0xFFFFu) << 16));             // dim0[31:16] | dim1[15:0]
  g1[3] = (int)(((tensor_d1 >> 16) & 0xFFFFu)
        | ((tile_d0 & 0xFFFFu) << 16));               // dim1[31:16] | tile_dim0
  g1[4] = (int)(tile_d1 & 0xFFFFu);                   // tile_dim1, tile_dim2=0
  g1[5] = (int)stride0;                               // tensor_dim0_stride[31:0]
  g1[6] = 0;                                          // stride0[47:32], dim1_stride lo
  g1[7] = 0;
  i32x4 z4 = {0, 0, 0, 0};
  i32x8 z8 = {0, 0, 0, 0, 0, 0, 0, 0};
  __builtin_amdgcn_tensor_load_to_lds(g0, g1, z4, z4, z8, 0);
}

static __device__ __forceinline__ unsigned lds_off_of(const void* p) {
  // Generic pointers into LDS carry the LDS offset in the low 32 bits.
  return (unsigned)(uintptr_t)p;
}

// ---------------------------------------------------------------------------
// Kernel 1: Wo (f32) -> WoH (f16)
// ---------------------------------------------------------------------------
__global__ __launch_bounds__(256) void wo_prep(const float* __restrict__ Wo,
                                               _Float16* __restrict__ WoH) {
  int i = blockIdx.x * 256 + threadIdx.x;
  if (i < CF_ * CF_) WoH[i] = (_Float16)Wo[i];
}

// ---------------------------------------------------------------------------
// Kernel 2: k = Wk @ feat + bk  -> kw window layout (f16)
//           v = feat            -> vwT transposed window layout (f16)
// One block per lo-res pixel (b,t,y,x). 128 threads = Cqk channels.
// ---------------------------------------------------------------------------
__global__ __launch_bounds__(128) void kv_kernel(const float* __restrict__ feat,
                                                 const float* __restrict__ Wk,
                                                 const float* __restrict__ bk,
                                                 _Float16* __restrict__ kw,
                                                 _Float16* __restrict__ vwT) {
  int pid = blockIdx.x;                 // 0 .. 8191
  int b   = pid >> 12;
  int rem = pid & 4095;
  int t   = rem >> 10;
  int yx  = rem & 1023;
  int y   = yx >> 5;
  int x   = yx & 31;

  __shared__ __align__(16) float fv[CF_];
  int tid = threadIdx.x;

  size_t fbase = (size_t)b * (CF_ * T_ * HL_ * WL_) + (size_t)t * 1024 + yx;
  #pragma unroll
  for (int c = tid; c < CF_; c += 128) fv[c] = feat[fbase + (size_t)c * 4096];
  __syncthreads();

  int tn = t >> 1, tt = t & 1;
  int hn = y >> 3, yy = y & 7;
  int wn = x >> 3, xx = x & 7;
  int win = (tn * 4 + hn) * 4 + wn;
  int key = tt * 64 + yy * 8 + xx;
  int whb = (b * NWIN_ + win) * NH_;

  // k projection: one channel per thread
  float acc = bk[tid];
  const float4* wrow = (const float4*)(Wk + (size_t)tid * CF_);
  const float4* fv4  = (const float4*)fv;
  #pragma unroll 8
  for (int c = 0; c < CF_ / 4; ++c) {
    float4 wv = wrow[c];
    float4 xv = fv4[c];
    acc += wv.x * xv.x + wv.y * xv.y + wv.z * xv.z + wv.w * xv.w;
  }
  int head = tid >> 5, dq = tid & 31;
  kw[(((size_t)(whb + head)) * KKEYS_ + key) * DQK_ + dq] = (_Float16)acc;

  // v transpose+convert: vwT[wh][dvi][key]
  #pragma unroll
  for (int c = tid; c < CF_; c += 128) {
    int hd = c / DV_, dvi = c % DV_;
    vwT[(((size_t)(whb + hd)) * DV_ + dvi) * KKEYS_ + key] = (_Float16)fv[c];
  }
}

// ---------------------------------------------------------------------------
// Kernel 3: windowed cross attention. One WG per (b, win, head). 4 waves.
// q computed on-the-fly from rgb (Ci=3). WMMA for S=q.k^T and O=P.V.
// Output scattered into X[pixel][Cf] (f16).
// ---------------------------------------------------------------------------
__global__ __launch_bounds__(128) void attn_kernel(const float* __restrict__ rgb,
                                                   const float* __restrict__ Wq,
                                                   const float* __restrict__ bq,
                                                   const _Float16* __restrict__ kw,
                                                   const _Float16* __restrict__ vwT,
                                                   _Float16* __restrict__ X) {
  int wh   = blockIdx.x;           // 0..255
  int head = wh & (NH_ - 1);
  int win  = (wh >> 2) & (NWIN_ - 1);
  int b    = wh >> 7;
  int wn = win & 3, hn = (win >> 2) & 3, tn = win >> 4;

  __shared__ __align__(32) _Float16 kw_s[KKEYS_ * DQK_];     // 8 KB
  __shared__ __align__(32) _Float16 vwT_s[DV_ * KKEYS_];     // 24 KB
  __shared__ __align__(32) _Float16 P_s[4][KKEYS_ * 16];     // 16 KB (per-wave P^T)
  __shared__ float wq_s[DQK_ * CI_];
  __shared__ float bq_s[DQK_];

  int tid  = threadIdx.x;
  int lane = tid & 31;
  int wv   = tid >> 5;
  int m    = lane & 15;
  int hf   = lane >> 4;

  // Stage K and V^T tiles into LDS (16B vector copies)
  {
    const uint4* gk = (const uint4*)(kw + (size_t)wh * (KKEYS_ * DQK_));
    uint4* sk = (uint4*)kw_s;
    #pragma unroll
    for (int i = tid; i < KKEYS_ * DQK_ / 8; i += 128) sk[i] = gk[i];
    const uint4* gv = (const uint4*)(vwT + (size_t)wh * (DV_ * KKEYS_));
    uint4* sv = (uint4*)vwT_s;
    #pragma unroll
    for (int i = tid; i < DV_ * KKEYS_ / 8; i += 128) sv[i] = gv[i];
    for (int i = tid; i < DQK_ * CI_; i += 128)
      wq_s[i] = Wq[(size_t)(head * DQK_ + i / CI_) * CI_ + (i % CI_)];
    if (tid < DQK_) bq_s[tid] = bq[head * DQK_ + tid];
  }
  __syncthreads();

  const float scale = 0.17677669529663687f;  // 1/sqrt(32)
  _Float16* Pw = &P_s[wv][0];

  for (int ch = wv; ch < QROWS_ / 16; ch += 4) {
    int row0 = ch * 16;

    // ---- build q A-fragment on the fly (scale folded in) ----
    int rr  = row0 + m;
    int tt  = rr >> 10, remq = rr & 1023, yy = remq >> 5, xq = remq & 31;
    int t   = tn * 2 + tt;
    int hh  = hn * 32 + yy;
    int ww  = wn * 32 + xq;
    size_t rgbb = (size_t)b * (CI_ * THW_) + (size_t)t * 16384 + hh * 128 + ww;
    float r0 = rgb[rgbb];
    float r1 = rgb[rgbb + THW_];
    float r2 = rgb[rgbb + 2 * THW_];

    v16h aq;
    #pragma unroll
    for (int e = 0; e < 16; ++e) {
      int g = e >> 3, j = e & 7;
      int k = hf * 16 + g * 8 + j;
      float qv = bq_s[k] + wq_s[k * 3 + 0] * r0 + wq_s[k * 3 + 1] * r1
                         + wq_s[k * 3 + 2] * r2;
      aq[e] = (_Float16)(qv * scale);
    }

    // ---- S = q . k^T : 8 WMMAs (K=32 in one shot) ----
    v8f sacc[8];
    #pragma unroll
    for (int nt = 0; nt < 8; ++nt) {
      v16h bk = *(const v16h*)&kw_s[(nt * 16 + m) * DQK_ + hf * 16];
      sacc[nt] = __builtin_amdgcn_wmma_f32_16x16x32_f16(
          false, aq, false, bk, (short)0, v8f_zero(), false, false);
    }

    // ---- softmax over 128 keys (rows split: lanes0-15 = M 0-7, lanes16-31 = M 8-15)
    float rsum[8];
    #pragma unroll
    for (int r = 0; r < 8; ++r) {
      float mx = sacc[0][r];
      #pragma unroll
      for (int nt = 1; nt < 8; ++nt) mx = fmaxf(mx, sacc[nt][r]);
      #pragma unroll
      for (int off = 1; off < 16; off <<= 1)
        mx = fmaxf(mx, __shfl_xor(mx, off, 32));
      float sm = 0.f;
      #pragma unroll
      for (int nt = 0; nt < 8; ++nt) {
        float p = __expf(sacc[nt][r] - mx);
        sacc[nt][r] = p;
        sm += p;
      }
      #pragma unroll
      for (int off = 1; off < 16; off <<= 1)
        sm += __shfl_xor(sm, off, 32);
      rsum[r] = 1.0f / sm;
    }

    // ---- pack P^T into LDS (b128 stores): P_s[key][qrow] ----
    #pragma unroll
    for (int nt = 0; nt < 8; ++nt) {
      v8h pk;
      #pragma unroll
      for (int r = 0; r < 8; ++r) pk[r] = (_Float16)(sacc[nt][r] * rsum[r]);
      int key = nt * 16 + m;
      *(v8h*)&Pw[key * 16 + hf * 8] = pk;
    }

    // ---- A-fragments of P (gathered from P^T, reused across all dv tiles) ----
    v16h ap[4];
    #pragma unroll
    for (int kk = 0; kk < 4; ++kk) {
      #pragma unroll
      for (int e = 0; e < 16; ++e) {
        int g = e >> 3, j = e & 7;
        int kidx = kk * 32 + hf * 16 + g * 8 + j;
        ap[kk][e] = Pw[kidx * 16 + m];
      }
    }

    // ---- output pixel addresses for this chunk ----
    size_t px[8];
    #pragma unroll
    for (int r = 0; r < 8; ++r) {
      int qr = hf * 8 + r;
      int rr2 = row0 + qr;
      int tt2 = rr2 >> 10, rem2 = rr2 & 1023, yy2 = rem2 >> 5, xx2 = rem2 & 31;
      px[r] = (size_t)b * THW_ + (size_t)(tn * 2 + tt2) * 16384
            + (hn * 32 + yy2) * 128 + (wn * 32 + xx2);
    }

    // ---- O = P . V : 24 WMMAs ----
    #pragma unroll
    for (int dvt = 0; dvt < 6; ++dvt) {
      v8f acc = v8f_zero();
      #pragma unroll
      for (int kk = 0; kk < 4; ++kk) {
        v16h bv = *(const v16h*)&vwT_s[(dvt * 16 + m) * KKEYS_ + kk * 32 + hf * 16];
        acc = __builtin_amdgcn_wmma_f32_16x16x32_f16(
            false, ap[kk], false, bv, (short)0, acc, false, false);
      }
      int c = head * DV_ + dvt * 16 + m;
      #pragma unroll
      for (int r = 0; r < 8; ++r)
        X[px[r] * CF_ + c] = (_Float16)acc[r];
    }
  }
}

// ---------------------------------------------------------------------------
// Kernel 4: Out = WoH @ X^T + bo   (the 38.7 GFLOP GEMM)
// WG tile: 128 out-channels x 128 pixels, K=384 in 32-chunks. Double-buffered
// LDS staging via the Tensor Data Mover (wave 0 issues 2-D tile DMAs; all 8
// waves run WMMAs on the other buffer). 96 WMMAs per wave.
// ---------------------------------------------------------------------------
__global__ __launch_bounds__(256) void out_kernel(const _Float16* __restrict__ X,
                                                  const _Float16* __restrict__ WoH,
                                                  const float* __restrict__ bo,
                                                  float* __restrict__ out) {
  __shared__ __align__(32) _Float16 wo_s[2][128 * 32];   // 2 x 8 KB
  __shared__ __align__(32) _Float16 x_s[2][128 * 32];    // 2 x 8 KB

  int tid  = threadIdx.x;
  int lane = tid & 31;
  int wv   = tid >> 5;
  int m    = lane & 15;
  int hf   = lane >> 4;
  int pbase = blockIdx.x * 128;
  int obase = blockIdx.y * 128;
  const int NK = CF_ / 32;     // 12

  v8f acc[8];
  #pragma unroll
  for (int pt = 0; pt < 8; ++pt) acc[pt] = v8f_zero();

  // Prologue: wave 0 kicks off the DMA for K-slice 0 into buffer 0.
  if (wv == 0) {
    tdm_load_tile_f16(&WoH[(size_t)obase * CF_ + 0], lds_off_of(&wo_s[0][0]),
                      32, 128, CF_, CF_, CF_);
    tdm_load_tile_f16(&X[(size_t)pbase * CF_ + 0], lds_off_of(&x_s[0][0]),
                      32, 128, CF_, NPIX_, CF_);
  }

  for (int kk = 0; kk < NK; ++kk) {
    int buf = kk & 1;
    __syncthreads();   // everyone done reading buf^1 before it is overwritten
    if (wv == 0) {
      if (kk + 1 < NK) {
        tdm_load_tile_f16(&WoH[(size_t)obase * CF_ + (kk + 1) * 32],
                          lds_off_of(&wo_s[buf ^ 1][0]), 32, 128, CF_, CF_, CF_);
        tdm_load_tile_f16(&X[(size_t)pbase * CF_ + (kk + 1) * 32],
                          lds_off_of(&x_s[buf ^ 1][0]), 32, 128, CF_, NPIX_, CF_);
        __builtin_amdgcn_s_wait_tensorcnt(2);   // slice kk done; kk+1 in flight
      } else {
        __builtin_amdgcn_s_wait_tensorcnt(0);   // last slice done
      }
    }
    __syncthreads();   // broadcast readiness of buf

    // A fragment: Wo rows wv*16 + m, two contiguous 8-half runs
    v16h a;
    {
      const _Float16* pA = &wo_s[buf][(wv * 16 + m) * 32];
      v8h lo = *(const v8h*)(pA + hf * 16);
      v8h hi = *(const v8h*)(pA + hf * 16 + 8);
      #pragma unroll
      for (int i = 0; i < 8; ++i) { a[i] = lo[i]; a[8 + i] = hi[i]; }
    }
    #pragma unroll
    for (int pt = 0; pt < 8; ++pt) {
      v16h bx = *(const v16h*)&x_s[buf][(pt * 16 + m) * 32 + hf * 16];
      acc[pt] = __builtin_amdgcn_wmma_f32_16x16x32_f16(
          false, a, false, bx, (short)0, acc[pt], false, false);
    }
  }

  // Epilogue: bias + coalesced f32 stores into [B][Cf][T][H][W]
  #pragma unroll
  for (int r = 0; r < 8; ++r) {
    int o = obase + wv * 16 + hf * 8 + r;
    float bias = bo[o];
    #pragma unroll
    for (int pt = 0; pt < 8; ++pt) {
      int p  = pbase + pt * 16 + m;
      int bb = p >> 16;
      int mm = p & (THW_ - 1);
      out[((size_t)(bb * CF_ + o)) * THW_ + mm] = acc[pt][r] + bias;
    }
  }
}

// ---------------------------------------------------------------------------
extern "C" void kernel_launch(void* const* d_in, const int* in_sizes, int n_in,
                              void* d_out, int out_size, void* d_ws, size_t ws_size,
                              hipStream_t stream) {
  const float* rgb  = (const float*)d_in[0];
  const float* feat = (const float*)d_in[1];
  const float* Wq   = (const float*)d_in[2];
  const float* bq   = (const float*)d_in[3];
  const float* Wk   = (const float*)d_in[4];
  const float* bk   = (const float*)d_in[5];
  const float* Wo   = (const float*)d_in[6];
  const float* bo   = (const float*)d_in[7];
  float* out = (float*)d_out;

  char* ws = (char*)d_ws;
  _Float16* WoH = (_Float16*)(ws + OFF_WOH);
  _Float16* kw  = (_Float16*)(ws + OFF_KW);
  _Float16* vwT = (_Float16*)(ws + OFF_VWT);
  _Float16* X   = (_Float16*)(ws + OFF_X);

  wo_prep  <<<(CF_ * CF_ + 255) / 256, 256, 0, stream>>>(Wo, WoH);
  kv_kernel<<<B_ * T_ * HL_ * WL_, 128, 0, stream>>>(feat, Wk, bk, kw, vwT);
  attn_kernel<<<B_ * NWIN_ * NH_, 128, 0, stream>>>(rgb, Wq, bq, kw, vwT, X);
  out_kernel<<<dim3(NPIX_ / 128, CF_ / 128), 256, 0, stream>>>(X, WoH, bo, out);
}
